// TseGen_47553877901969
// MI455X (gfx1250) — matrix operand order repaired and checked by reading, not compiled, not run on hardware
//
#include <hip/hip_runtime.h>
#include <math.h>

// ---------------------------------------------------------------- types
typedef __attribute__((ext_vector_type(16))) __bf16 v16bf;
typedef __attribute__((ext_vector_type(8)))  float  v8f;

__device__ __forceinline__ v8f wmma_bf16(v16bf a, v16bf b, v8f c) {
  // D = A(16x32 bf16) * B(32x16 bf16) + C(16x16 f32)
  return __builtin_amdgcn_wmma_f32_16x16x32_bf16(false, a, false, b, (short)0, c, false, false);
}

__device__ __forceinline__ float gelu_f(float x) {
  const float c = 0.7978845608028654f; // sqrt(2/pi)
  return 0.5f * x * (1.0f + tanhf(c * (x + 0.044715f * x * x * x)));
}
__device__ __forceinline__ float sigmoid_f(float x) { return 1.0f / (1.0f + __expf(-x)); }

// pack 8 consecutive f32 (16B-aligned) into bf16 fragment elements [e0..e0+7]
__device__ __forceinline__ void pack8(v16bf& a, int e0, const float* __restrict__ p) {
  float4 lo = *(const float4*)p;
  float4 hi = *(const float4*)(p + 4);
  a[e0 + 0] = (__bf16)lo.x; a[e0 + 1] = (__bf16)lo.y;
  a[e0 + 2] = (__bf16)lo.z; a[e0 + 3] = (__bf16)lo.w;
  a[e0 + 4] = (__bf16)hi.x; a[e0 + 5] = (__bf16)hi.y;
  a[e0 + 6] = (__bf16)hi.z; a[e0 + 7] = (__bf16)hi.w;
}

// ---------------------------------------------------------------- generic WMMA GEMM
// C[M,N] = act( A[M,K] @ W[K,N] + bias[N] (+ edge node terms) )
// block = 128 threads (4 waves); each wave one 16x16 tile; block covers 64(M) x 16(N).
// GUARD=false requires: K%32==0, M%64==0, N%16==0, lda%4==0, A 16B-aligned.
template<bool GUARD>
__global__ __launch_bounds__(128) void gemm_kernel(
    const float* __restrict__ A, const float* __restrict__ W,
    const float* __restrict__ bias, float* __restrict__ C,
    int M, int K, int N, int lda, int ldw, int ldc, int act,
    const float* __restrict__ addI, const float* __restrict__ addJ)
{
  __shared__ __bf16 Wt[16][32];  // transposed W tile: [n_local][k_local]
  const int tid  = threadIdx.x;
  const int wave = tid >> 5;
  const int lane = tid & 31;
  const int hi   = lane >> 4;
  const int l15  = lane & 15;
  const int m0   = blockIdx.x * 64 + wave * 16;
  const int n0   = blockIdx.y * 16;

  v8f acc = {};
  const int arow = GUARD ? min(m0 + l15, M - 1) : (m0 + l15);
  const float* Arow = A + (size_t)arow * lda;

  for (int k0 = 0; k0 < K; k0 += 32) {
    for (int s = tid; s < 16 * 32; s += 128) {
      int n = s >> 5, kk = s & 31;
      if (GUARD) {
        float wv = 0.f;
        if (k0 + kk < K && n0 + n < N) wv = W[(size_t)(k0 + kk) * ldw + n0 + n];
        Wt[n][kk] = (__bf16)wv;
      } else {
        Wt[n][kk] = (__bf16)W[(size_t)(k0 + kk) * ldw + n0 + n];
      }
    }
    __syncthreads();
    v16bf a, b;
    if (GUARD) {
#pragma unroll
      for (int e = 0; e < 8; ++e) {  // ISA 16-bit A layout
        int k1 = k0 + 8 * hi + e;
        int k2 = k0 + 16 + 8 * hi + e;
        a[e]     = (__bf16)((k1 < K) ? Arow[k1] : 0.f);
        a[8 + e] = (__bf16)((k2 < K) ? Arow[k2] : 0.f);
      }
    } else {
      const float* Ab = Arow + k0 + 8 * hi;
      pack8(a, 0, Ab);        // K = k0 + 8*hi + {0..7}
      pack8(a, 8, Ab + 16);   // K = k0 + 16 + 8*hi + {0..7}
    }
#pragma unroll
    for (int e = 0; e < 16; ++e) b[e] = Wt[l15][8 * hi + (e & 7) + ((e >> 3) << 4)];  // B: lane n, K = 8*hi + (e&7) + 16*(e>>3)
    acc = wmma_bf16(a, b, acc);
    __syncthreads();
  }
#pragma unroll
  for (int r = 0; r < 8; ++r) {   // D layout: m = r + 8*hi, n = lane&15
    int mm = m0 + r + 8 * hi;
    int nn = n0 + l15;
    if (!GUARD || (mm < M && nn < N)) {
      float v = acc[r];
      if (bias) v += bias[nn];
      if (addI) // edge mode: row = b*65536 + i*256 + j  (N==64 feature buffers)
        v += addI[(size_t)(mm >> 8) * N + nn] +
             addJ[(size_t)((mm >> 16) * 256 + (mm & 255)) * N + nn];
      if (act == 1) v = gelu_f(v);
      else if (act == 2) v = sigmoid_f(v);
      C[(size_t)mm * ldc + nn] = v;
    }
  }
}

// ---------------------------------------------------------------- attention softmax
// grid (N, B), block N(=256). writes attn[b,h,i,j].
__global__ void attn_kernel(const float* __restrict__ q, const float* __restrict__ k,
                            const float* __restrict__ biasb, const float* __restrict__ mask,
                            float* __restrict__ attn, int N, int D, int dh, float scale)
{
  __shared__ float qs[128];
  __shared__ float red[8];
  const int b = blockIdx.y, i = blockIdx.x, j = threadIdx.x;
  for (int s = j; s < D; s += blockDim.x) qs[s] = q[(size_t)(b * N + i) * D + s];
  __syncthreads();
  const float* krow = k + (size_t)(b * N + j) * D;
  const bool ok = mask[b * N + j] > 0.f;
  float logit[4];
#pragma unroll
  for (int h = 0; h < 4; ++h) {
    float d = 0.f;
    for (int t = 0; t < dh; ++t) d += qs[h * dh + t] * krow[h * dh + t];
    d = d * scale + biasb[((size_t)(b * N + i) * N + j) * 4 + h];
    logit[h] = ok ? d : -1e9f;
  }
  const int nw = blockDim.x >> 5;
#pragma unroll
  for (int h = 0; h < 4; ++h) {
    float m = logit[h];
    for (int off = 16; off; off >>= 1) m = fmaxf(m, __shfl_xor(m, off, 32));
    if ((j & 31) == 0) red[j >> 5] = m;
    __syncthreads();
    float bm = red[0];
    for (int w = 1; w < nw; ++w) bm = fmaxf(bm, red[w]);
    __syncthreads();
    float e = __expf(logit[h] - bm);
    float s = e;
    for (int off = 16; off; off >>= 1) s += __shfl_xor(s, off, 32);
    if ((j & 31) == 0) red[j >> 5] = s;
    __syncthreads();
    float bs = 0.f;
    for (int w = 0; w < nw; ++w) bs += red[w];
    __syncthreads();
    attn[((size_t)(b * 4 + h) * N + i) * N + j] = e / bs;
  }
}

// ---------------------------------------------------------------- residual (gated) LayerNorm
__global__ void resid_ln_kernel(const float* __restrict__ x, const float* __restrict__ m,
                                const float* __restrict__ gate, const float* __restrict__ g,
                                const float* __restrict__ beta, float* __restrict__ out, int D)
{
  __shared__ float red[8];
  const int row = blockIdx.x, t = threadIdx.x;
  const size_t idx = (size_t)row * D + t;
  float val = x[idx] + (gate ? gate[idx] * m[idx] : m[idx]);
  const int nw = blockDim.x >> 5;
  float s = val;
  for (int off = 16; off; off >>= 1) s += __shfl_xor(s, off, 32);
  if ((t & 31) == 0) red[t >> 5] = s;
  __syncthreads();
  float tot = 0.f;
  for (int w = 0; w < nw; ++w) tot += red[w];
  const float mean = tot / D;
  __syncthreads();
  float d2 = (val - mean) * (val - mean);
  for (int off = 16; off; off >>= 1) d2 += __shfl_xor(d2, off, 32);
  if ((t & 31) == 0) red[t >> 5] = d2;
  __syncthreads();
  float var = 0.f;
  for (int w = 0; w < nw; ++w) var += red[w];
  var /= D;
  out[idx] = (val - mean) * rsqrtf(var + 1e-5f) * g[t] + beta[t];
}

// ---------------------------------------------------------------- fused edge update
// es_out = LN( es + sigmoid(es@weg + beg) * (tmpE@we2), lng, lnb ). ES=64.
__global__ __launch_bounds__(128) void edge_update_kernel(
    const float* __restrict__ es, const float* __restrict__ tmpE,
    const float* __restrict__ we2, const float* __restrict__ weg,
    const float* __restrict__ beg, const float* __restrict__ lng,
    const float* __restrict__ lnb, float* __restrict__ out)
{
  __shared__ __bf16 W2t[64][64];
  __shared__ __bf16 Wgt[64][64];
  __shared__ float sums[16], sumsq[16];
  const int tid = threadIdx.x, wave = tid >> 5, lane = tid & 31;
  const int hi = lane >> 4, l15 = lane & 15;
  const int m0 = blockIdx.x * 16, n0 = wave * 16;
  for (int s = tid; s < 64 * 64; s += 128) {
    int n = s >> 6, kk = s & 63;
    W2t[n][kk] = (__bf16)we2[kk * 64 + n];
    Wgt[n][kk] = (__bf16)weg[kk * 64 + n];
  }
  if (tid < 16) { sums[tid] = 0.f; sumsq[tid] = 0.f; }
  __syncthreads();
  v8f amsg = {}, age = {};
  const float* esr = es   + (size_t)(m0 + l15) * 64;
  const float* ter = tmpE + (size_t)(m0 + l15) * 64;
#pragma unroll
  for (int k0 = 0; k0 < 64; k0 += 32) {
    v16bf ae, at, b2, bg;
    pack8(ae, 0, esr + k0 + 8 * hi);  pack8(ae, 8, esr + k0 + 16 + 8 * hi);
    pack8(at, 0, ter + k0 + 8 * hi);  pack8(at, 8, ter + k0 + 16 + 8 * hi);
#pragma unroll
    for (int e = 0; e < 16; ++e) {
      b2[e] = W2t[n0 + l15][k0 + 8 * hi + (e & 7) + ((e >> 3) << 4)];
      bg[e] = Wgt[n0 + l15][k0 + 8 * hi + (e & 7) + ((e >> 3) << 4)];
    }
    amsg = wmma_bf16(at, b2, amsg);
    age  = wmma_bf16(ae, bg, age);
  }
  const int nn = n0 + l15;
  float tv[8];
#pragma unroll
  for (int r = 0; r < 8; ++r) {
    int ml = r + 8 * hi;
    float ge = sigmoid_f(age[r] + beg[nn]);
    float t  = es[(size_t)(m0 + ml) * 64 + nn] + ge * amsg[r];
    tv[r] = t;
    atomicAdd(&sums[ml], t);
    atomicAdd(&sumsq[ml], t * t);
  }
  __syncthreads();
#pragma unroll
  for (int r = 0; r < 8; ++r) {
    int ml = r + 8 * hi;
    float mean = sums[ml] * (1.f / 64.f);
    float var  = sumsq[ml] * (1.f / 64.f) - mean * mean;
    out[(size_t)(m0 + ml) * 64 + nn] = (tv[r] - mean) * rsqrtf(var + 1e-5f) * lng[nn] + lnb[nn];
  }
}

// ---------------------------------------------------------------- fused edge gate
// g = sigmoid(concat(r,p)@egw + egb); out = g*r + (1-g)*p   (out may alias res)
__global__ __launch_bounds__(128) void edge_gate_kernel(
    const float* __restrict__ res, const float* __restrict__ pes,
    const float* __restrict__ egw, const float* __restrict__ egb,
    float* __restrict__ out)
{
  __shared__ __bf16 Wt[64][128];
  const int tid = threadIdx.x, wave = tid >> 5, lane = tid & 31;
  const int hi = lane >> 4, l15 = lane & 15;
  const int m0 = blockIdx.x * 16, n0 = wave * 16;
  for (int s = tid; s < 64 * 128; s += 128) {
    int n = s >> 7, kk = s & 127;
    Wt[n][kk] = (__bf16)egw[kk * 64 + n];
  }
  __syncthreads();
  v8f acc = {};
  const float* rr = res + (size_t)(m0 + l15) * 64;
  const float* pr = pes + (size_t)(m0 + l15) * 64;
#pragma unroll
  for (int k0 = 0; k0 < 128; k0 += 32) {
    v16bf a, b;
    const float* s1 = (k0 < 64) ? (rr + k0) : (pr + k0 - 64);        // chunk k0+8hi..+8
    const float* s2 = (k0 + 16 < 64) ? (rr + k0 + 16) : (pr + k0 - 48);
    pack8(a, 0, s1 + 8 * hi);
    pack8(a, 8, s2 + 8 * hi);
#pragma unroll
    for (int e = 0; e < 16; ++e) b[e] = Wt[n0 + l15][k0 + 8 * hi + (e & 7) + ((e >> 3) << 4)];
    acc = wmma_bf16(a, b, acc);
  }
  const int nn = n0 + l15;
  float rv[8], pv[8], gv[8];
#pragma unroll
  for (int r = 0; r < 8; ++r) {
    int mm = m0 + r + 8 * hi;
    rv[r] = res[(size_t)mm * 64 + nn];
    pv[r] = pes[(size_t)mm * 64 + nn];
    gv[r] = sigmoid_f(acc[r] + egb[nn]);
  }
  __syncthreads();  // all reads of res done before in-place writes
#pragma unroll
  for (int r = 0; r < 8; ++r) {
    int mm = m0 + r + 8 * hi;
    out[(size_t)mm * 64 + nn] = gv[r] * rv[r] + (1.f - gv[r]) * pv[r];
  }
}

// ---------------------------------------------------------------- small elementwise kernels
__global__ void concat_kernel(const float* __restrict__ a, const float* __restrict__ b,
                              float* __restrict__ out, int M, int D)
{
  int i = blockIdx.x * blockDim.x + threadIdx.x;
  if (i < M * 2 * D) {
    int m = i / (2 * D), c = i % (2 * D);
    out[i] = (c < D) ? a[(size_t)m * D + c] : b[(size_t)m * D + c - D];
  }
}

__global__ void node_combine_kernel(const float* __restrict__ g, const float* __restrict__ r,
                                    const float* __restrict__ p, const float* __restrict__ sigmas,
                                    const float* __restrict__ sigw, const float* __restrict__ sigb,
                                    float* __restrict__ out, int M, int D)
{
  int i = blockIdx.x * blockDim.x + threadIdx.x;
  if (i < M * D) {
    int m = i / D, c = i % D, b = m >> 8;
    out[i] = g[i] * r[i] + (1.f - g[i]) * p[i] + sigmas[b] * sigw[c] + sigb[c];
  }
}

// V[b,i,c,x] += sum_j mean_h(attn) * cbuf[b,i,j,c] * (ts[b,i]-ts[b,j])[x]
__global__ void vacc_kernel(const float* __restrict__ attn, const float* __restrict__ cbuf,
                            const float* __restrict__ ts, float* __restrict__ V, int N)
{
  __shared__ float vac[48];
  const int b = blockIdx.y, i = blockIdx.x, j = threadIdx.x;
  if (j < 48) vac[j] = 0.f;
  __syncthreads();
  float am = 0.f;
#pragma unroll
  for (int h = 0; h < 4; ++h) am += attn[((size_t)(b * 4 + h) * N + i) * N + j];
  am *= 0.25f;
  float rel[3];
#pragma unroll
  for (int x = 0; x < 3; ++x) rel[x] = ts[(b * N + i) * 3 + x] - ts[(b * N + j) * 3 + x];
  const float* cr = cbuf + ((size_t)(b * N + i) * N + j) * 16;
  for (int cc = 0; cc < 16; ++cc) {
    float cm = cr[cc] * am;
#pragma unroll
    for (int x = 0; x < 3; ++x) {
      float val = cm * rel[x];
      for (int off = 16; off; off >>= 1) val += __shfl_xor(val, off, 32);
      if ((j & 31) == 0) atomicAdd(&vac[cc * 3 + x], val);
    }
  }
  __syncthreads();
  if (j < 48) V[(size_t)(b * N + i) * 48 + j] += vac[j];
}

__global__ void energy_kernel(const float* __restrict__ h, const float* __restrict__ w0,
                              const float* __restrict__ b0, const float* __restrict__ mask,
                              float* __restrict__ out, int N, int D)
{
  __shared__ float red[8];
  const int b = blockIdx.x, i = threadIdx.x;
  const float* hr = h + (size_t)(b * N + i) * D;
  float d = 0.f;
  for (int c = 0; c < D; ++c) d += hr[c] * w0[c];
  d = (d + b0[0]) * mask[b * N + i];
  for (int off = 16; off; off >>= 1) d += __shfl_xor(d, off, 32);
  if ((i & 31) == 0) red[i >> 5] = d;
  __syncthreads();
  if (i == 0) {
    float s = 0.f;
    for (int w = 0; w < (blockDim.x >> 5); ++w) s += red[w];
    out[b] = s;
  }
}

__global__ void forces_kernel(const float* __restrict__ V, const float* __restrict__ w1,
                              const float* __restrict__ mask, float* __restrict__ out, int BN)
{
  int i = blockIdx.x * blockDim.x + threadIdx.x;
  if (i < BN) {
    const float* vr = V + (size_t)i * 48;
    float f0 = 0.f, f1 = 0.f, f2 = 0.f;
    for (int c = 0; c < 16; ++c) {
      f0 += vr[c * 3 + 0] * w1[c];
      f1 += vr[c * 3 + 1] * w1[c];
      f2 += vr[c * 3 + 2] * w1[c];
    }
    float mk = mask[i];
    out[2 + i * 3 + 0] = f0 * mk;
    out[2 + i * 3 + 1] = f1 * mk;
    out[2 + i * 3 + 2] = f2 * mk;
  }
}

// ================================================================ host side
namespace {

constexpr int BATCH = 2, NN_ = 256, BN_ = 512, DS_ = 128, ES_ = 64;
constexpr int NEDGE = BATCH * NN_ * NN_;  // 131072 edge rows

// ---- parameter flat offsets (jax tree_leaves order: sorted dict keys) ----
constexpr size_t P_DEC_IN_B = 0, P_DEC_IN_W = 32;
constexpr size_t P_DEC_L = 4128, DEC_STRIDE = 5440;
constexpr size_t DL_LNB = 0, DL_LNG = 32, DL_WB = 64, DL_WC = 320, DL_WK = 1344,
                 DL_WO = 2368, DL_WQ = 3392, DL_WV = 4416;
constexpr size_t P_EG_B = 25888, P_EG_W = 25952;
constexpr size_t P_OUT0_B = 36208, P_OUT0_W = 36209, P_OUT1_W = 36241;
constexpr size_t PREF_P = 36257, PREF_R = 774225;  // 'p_*' block, 'r_*' block
// prefix-relative
constexpr size_t E_EMB_B = 0, E_EMB_W = 64, LAYERS = 400, LAYER_STRIDE = 183712;
constexpr size_t X_EMB_B = 735248, X_EMB_W = 735376;
// encoder-layer-relative (sorted leaf order)
constexpr size_t L_BEG = 0, L_BGS = 64, L_LN1B = 224, L_LN1G = 352, L_LN2B = 480,
                 L_LN2G = 608, L_LNEB = 736, L_LNEG = 800, L_W1 = 896, L_W2 = 33664,
                 L_WB = 66432, L_WE1 = 66720, L_WE2 = 70816, L_WEG = 74912,
                 L_WGS = 79264, L_WK = 99744, L_WNI = 116128, L_WNJ = 124320,
                 L_WO = 132512, L_WQ = 149920, L_WV = 166304;
constexpr size_t P_SIG_B = 1512193, P_SIG_W = 1512321, P_XG_B = 1512449, P_XG_W = 1512577;

// ---- workspace layout ----
struct WS {
  float *esA, *esB, *tmpE, *attn, *biasb;
  float *xsA, *xsB, *q, *k, *v, *gs, *msg, *tmp1, *tmp2, *cat, *g, *ni, *nj, *xcomb;
  float *h, *dq, *dk, *dv, *dmsg, *dtmp, *V;
};

inline void gemm_fast(hipStream_t s, const float* A, const float* W, const float* bias, float* C,
                      int M, int K, int N, int lda, int ldw, int ldc, int act = 0,
                      const float* addI = nullptr, const float* addJ = nullptr)
{
  dim3 grid(M / 64, N / 16);
  gemm_kernel<false><<<grid, 128, 0, s>>>(A, W, bias, C, M, K, N, lda, ldw, ldc, act, addI, addJ);
}
inline void gemm_guard(hipStream_t s, const float* A, const float* W, const float* bias, float* C,
                       int M, int K, int N, int lda, int ldw, int ldc, int act = 0)
{
  dim3 grid((M + 63) / 64, (N + 15) / 16);
  gemm_kernel<true><<<grid, 128, 0, s>>>(A, W, bias, C, M, K, N, lda, ldw, ldc, act,
                                         nullptr, nullptr);
}

void run_encoder(hipStream_t s, const float* P, size_t pref, const float* atoms,
                 const float* adjs, const float* mask, float* xs, float* es, WS& w)
{
  // embeddings (ragged K -> guarded)
  gemm_guard(s, atoms, P + pref + X_EMB_W, P + pref + X_EMB_B, xs, BN_, 20, DS_, 20, DS_, DS_);
  gemm_guard(s, adjs,  P + pref + E_EMB_W, P + pref + E_EMB_B, es, NEDGE, 5, ES_, 5, ES_, ES_);
  for (int L = 0; L < 4; ++L) {
    const float* lp = P + pref + LAYERS + (size_t)L * LAYER_STRIDE;
    // projections
    gemm_fast(s, xs, lp + L_WQ,  nullptr,    w.q,  BN_, DS_, DS_, DS_, DS_, DS_);
    gemm_fast(s, xs, lp + L_WK,  nullptr,    w.k,  BN_, DS_, DS_, DS_, DS_, DS_);
    gemm_fast(s, xs, lp + L_WV,  nullptr,    w.v,  BN_, DS_, DS_, DS_, DS_, DS_);
    gemm_fast(s, xs, lp + L_WGS, lp + L_BGS, w.gs, BN_, DS_, DS_, DS_, DS_, DS_, 2);
    // edge attention bias [b,i,j,h]  (N=4 -> guarded)
    gemm_guard(s, es, lp + L_WB, nullptr, w.biasb, NEDGE, ES_, 4, ES_, 4, 4);
    // softmax
    attn_kernel<<<dim3(NN_, BATCH), NN_, 0, s>>>(w.q, w.k, w.biasb, mask, w.attn,
                                                 NN_, DS_, 32, 0.17677669529663687f);
    // msg = attn @ v, per (b,h)
    for (int b = 0; b < BATCH; ++b)
      for (int h = 0; h < 4; ++h)
        gemm_fast(s, w.attn + ((size_t)(b * 4 + h) * NN_) * NN_,
                  w.v + (size_t)b * NN_ * DS_ + h * 32, nullptr,
                  w.msg + (size_t)b * NN_ * DS_ + h * 32,
                  NN_, NN_, 32, NN_, DS_, DS_);
    gemm_fast(s, w.msg, lp + L_WO, nullptr, w.tmp2, BN_, DS_, DS_, DS_, DS_, DS_);
    resid_ln_kernel<<<BN_, DS_, 0, s>>>(xs, w.tmp2, w.gs, lp + L_LN1G, lp + L_LN1B, xs, DS_);
    // FFN
    gemm_fast(s, xs, lp + L_W1, nullptr, w.tmp1, BN_, DS_, 256, DS_, 256, 256, 1);
    gemm_fast(s, w.tmp1, lp + L_W2, nullptr, w.tmp2, BN_, 256, DS_, 256, DS_, DS_);
    resid_ln_kernel<<<BN_, DS_, 0, s>>>(xs, w.tmp2, nullptr, lp + L_LN2G, lp + L_LN2B, xs, DS_);
    // edge MLP
    gemm_fast(s, xs, lp + L_WNI, nullptr, w.ni, BN_, DS_, ES_, DS_, ES_, ES_);
    gemm_fast(s, xs, lp + L_WNJ, nullptr, w.nj, BN_, DS_, ES_, DS_, ES_, ES_);
    gemm_fast(s, es, lp + L_WE1, nullptr, w.tmpE, NEDGE, ES_, ES_, ES_, ES_, ES_, 1, w.ni, w.nj);
    edge_update_kernel<<<NEDGE / 16, 128, 0, s>>>(es, w.tmpE, lp + L_WE2, lp + L_WEG,
                                                  lp + L_BEG, lp + L_LNEG, lp + L_LNEB, es);
  }
}

} // namespace

extern "C" void kernel_launch(void* const* d_in, const int* in_sizes, int n_in,
                              void* d_out, int out_size, void* d_ws, size_t ws_size,
                              hipStream_t stream)
{
  (void)in_sizes; (void)n_in; (void)out_size; (void)ws_size;
  const float* ratoms   = (const float*)d_in[0];
  const float* patoms   = (const float*)d_in[1];
  const float* radjs    = (const float*)d_in[2];
  const float* padjs    = (const float*)d_in[3];
  const float* tscoords = (const float*)d_in[6];
  const float* sigmas   = (const float*)d_in[7];
  const float* masks    = (const float*)d_in[8];
  const float* P        = (const float*)d_in[9];
  float* out = (float*)d_out;

  float* f = (float*)d_ws;
  WS w;
  size_t o = 0;
  w.esA   = f + o; o += (size_t)NEDGE * ES_;
  w.esB   = f + o; o += (size_t)NEDGE * ES_;
  w.tmpE  = f + o; o += (size_t)NEDGE * ES_;
  w.attn  = f + o; o += (size_t)BATCH * 4 * NN_ * NN_;
  w.biasb = f + o; o += (size_t)NEDGE * 4;
  w.xsA   = f + o; o += BN_ * DS_;
  w.xsB   = f + o; o += BN_ * DS_;
  w.q     = f + o; o += BN_ * DS_;
  w.k     = f + o; o += BN_ * DS_;
  w.v     = f + o; o += BN_ * DS_;
  w.gs    = f + o; o += BN_ * DS_;
  w.msg   = f + o; o += BN_ * DS_;
  w.tmp1  = f + o; o += BN_ * 256;
  w.tmp2  = f + o; o += BN_ * DS_;
  w.cat   = f + o; o += BN_ * 256;
  w.g     = f + o; o += BN_ * DS_;
  w.ni    = f + o; o += BN_ * ES_;
  w.nj    = f + o; o += BN_ * ES_;
  w.xcomb = f + o; o += BN_ * DS_;
  w.h     = f + o; o += BN_ * 32;
  w.dq    = f + o; o += BN_ * 32;
  w.dk    = f + o; o += BN_ * 32;
  w.dv    = f + o; o += BN_ * 32;
  w.dmsg  = f + o; o += BN_ * 32;
  w.dtmp  = f + o; o += BN_ * 32;
  w.V     = f + o; o += BN_ * 48;

  // NB reference swap: 'r' params process product inputs, 'p' params process reactant inputs
  run_encoder(stream, P, PREF_R, patoms, padjs, masks, w.xsB, w.esB, w);  // -> p_xs/p_es
  run_encoder(stream, P, PREF_P, ratoms, radjs, masks, w.xsA, w.esA, w);  // -> r_xs/r_es

  // node gate + sigma embedding
  concat_kernel<<<(BN_ * 256 + 255) / 256, 256, 0, stream>>>(w.xsA, w.xsB, w.cat, BN_, DS_);
  gemm_fast(stream, w.cat, P + P_XG_W, P + P_XG_B, w.g, BN_, 256, DS_, 256, DS_, DS_, 2);
  node_combine_kernel<<<(BN_ * DS_ + 255) / 256, 256, 0, stream>>>(
      w.g, w.xsA, w.xsB, sigmas, P + P_SIG_W, P + P_SIG_B, w.xcomb, BN_, DS_);
  // edge gate (in place into esA)
  edge_gate_kernel<<<NEDGE / 16, 128, 0, stream>>>(w.esA, w.esB, P + P_EG_W, P + P_EG_B, w.esA);

  // decoder
  gemm_fast(stream, w.xcomb, P + P_DEC_IN_W, P + P_DEC_IN_B, w.h, BN_, DS_, 32, DS_, 32, 32);
  (void)hipMemsetAsync(w.V, 0, (size_t)BN_ * 48 * sizeof(float), stream);
  for (int L = 0; L < 4; ++L) {
    const float* dl = P + P_DEC_L + (size_t)L * DEC_STRIDE;
    gemm_fast(stream, w.h, dl + DL_WQ, nullptr, w.dq, BN_, 32, 32, 32, 32, 32);
    gemm_fast(stream, w.h, dl + DL_WK, nullptr, w.dk, BN_, 32, 32, 32, 32, 32);
    gemm_fast(stream, w.h, dl + DL_WV, nullptr, w.dv, BN_, 32, 32, 32, 32, 32);
    gemm_guard(stream, w.esA, dl + DL_WB, nullptr, w.biasb, NEDGE, ES_, 4, ES_, 4, 4);
    attn_kernel<<<dim3(NN_, BATCH), NN_, 0, stream>>>(w.dq, w.dk, w.biasb, masks, w.attn,
                                                      NN_, 32, 8, 0.35355339059327373f);
    for (int b = 0; b < BATCH; ++b)
      for (int h = 0; h < 4; ++h)   // N=8 -> guarded
        gemm_guard(stream, w.attn + ((size_t)(b * 4 + h) * NN_) * NN_,
                   w.dv + (size_t)b * NN_ * 32 + h * 8, nullptr,
                   w.dmsg + (size_t)b * NN_ * 32 + h * 8,
                   NN_, NN_, 8, NN_, 32, 32);
    gemm_fast(stream, w.dmsg, dl + DL_WO, nullptr, w.dtmp, BN_, 32, 32, 32, 32, 32);
    resid_ln_kernel<<<BN_, 32, 0, stream>>>(w.h, w.dtmp, nullptr, dl + DL_LNG, dl + DL_LNB,
                                            w.h, 32);
    gemm_fast(stream, w.esA, dl + DL_WC, nullptr, w.tmpE, NEDGE, ES_, 16, ES_, 16, 16);
    vacc_kernel<<<dim3(NN_, BATCH), NN_, 0, stream>>>(w.attn, w.tmpE, tscoords, w.V, NN_);
  }

  // outputs: energy [2], forces [2,256,3]
  energy_kernel<<<BATCH, NN_, 0, stream>>>(w.h, P + P_OUT0_W, P + P_OUT0_B, masks, out, NN_, 32);
  forces_kernel<<<(BN_ + 127) / 128, 128, 0, stream>>>(w.V, P + P_OUT1_W, masks, out, BN_);
}
